// Track_On2_37056977830558
// MI455X (gfx1250) — compile-verified, matrix operand order checked by reading, not run on hardware
//
#include <hip/hip_runtime.h>

#define Bx 1
#define Tx 16
#define Nq 256
#define Mm 12
#define Dd 256
#define HF 64
#define WF 64
#define Pp (HF*WF)

typedef __attribute__((ext_vector_type(16))) __bf16          v16bf;
typedef __attribute__((ext_vector_type(8)))  float           v8f;
typedef __attribute__((ext_vector_type(16))) unsigned short  v16u;
typedef __attribute__((ext_vector_type(8)))  unsigned short  v8u;
typedef __attribute__((ext_vector_type(4)))  unsigned int    v4ui;
typedef __attribute__((ext_vector_type(8)))  int             v8ii;
typedef __attribute__((ext_vector_type(4)))  int             v4ii;

#if __has_builtin(__builtin_amdgcn_tensor_load_to_lds)
#define USE_TDM 1
#endif

__device__ __forceinline__ unsigned short f32_to_bf16(float f) {
    unsigned int x = __builtin_bit_cast(unsigned int, f);
    unsigned int r = x + 0x7FFFu + ((x >> 16) & 1u);   // round-to-nearest-even
    return (unsigned short)(r >> 16);
}

__device__ __forceinline__ int clampi(int v, int lo, int hi) {
    return v < lo ? lo : (v > hi ? hi : v);
}

// ---------------------------------------------------------------- zero state
__global__ void zero_state_kernel(float* memb, float* validf) {
    int idx = blockIdx.x * blockDim.x + threadIdx.x;
    if (idx < Nq * Mm * Dd) memb[idx] = 0.0f;
    if (idx < Nq * Mm)      validf[idx] = 0.0f;
}

// ------------------------------------------------------- bilinear grid sample
__global__ void grid_sample_kernel(const float* __restrict__ ffu,
                                   const float* __restrict__ qxy,
                                   const int*   __restrict__ qt,
                                   float* __restrict__ q_init) {
    int n = blockIdx.x;
    int d = threadIdx.x;
    float gx = qxy[n * 2 + 0] * 2.0f - 1.0f;
    float gy = qxy[n * 2 + 1] * 2.0f - 1.0f;
    float x = ((gx + 1.0f) * (float)WF - 1.0f) * 0.5f;
    float y = ((gy + 1.0f) * (float)HF - 1.0f) * 0.5f;
    float x0f = floorf(x), y0f = floorf(y);
    float wx = x - x0f, wy = y - y0f;
    int x0 = clampi((int)x0f,     0, WF - 1);
    int x1 = clampi((int)x0f + 1, 0, WF - 1);
    int y0 = clampi((int)y0f,     0, HF - 1);
    int y1 = clampi((int)y0f + 1, 0, HF - 1);
    const float* base = ffu + (size_t)qt[n] * Pp * Dd;
    float v00 = base[(size_t)(y0 * WF + x0) * Dd + d];
    float v01 = base[(size_t)(y0 * WF + x1) * Dd + d];
    float v10 = base[(size_t)(y1 * WF + x0) * Dd + d];
    float v11 = base[(size_t)(y1 * WF + x1) * Dd + d];
    q_init[n * Dd + d] = v00 * (1 - wx) * (1 - wy) + v01 * wx * (1 - wy)
                       + v10 * (1 - wx) * wy       + v11 * wx * wy;
}

// ------------------------------------------- per-frame query update + mem roll
__global__ void qupdate_kernel(const float* __restrict__ q_init,
                               float* __restrict__ memb,
                               float* __restrict__ validf,
                               const int* __restrict__ qt,
                               int t,
                               unsigned short* __restrict__ qn) {
    int n = blockIdx.x;
    int d = threadIdx.x;
    __shared__ float red[Dd];

    float vals[Mm];
    float cnt = 0.0f, mr = 0.0f;
    #pragma unroll
    for (int m = 0; m < Mm; ++m) {
        vals[m] = validf[n * Mm + m];
        cnt += vals[m];
        mr  += memb[((size_t)n * Mm + m) * Dd + d] * vals[m];
    }
    float q = q_init[n * Dd + d] + mr / fmaxf(cnt, 1.0f);

    red[d] = q * q;
    __syncthreads();
    for (int s = Dd / 2; s > 0; s >>= 1) {
        if (d < s) red[d] += red[d + s];
        __syncthreads();
    }
    float inv = 1.0f / fmaxf(sqrtf(red[0]), 1e-12f);
    qn[n * Dd + d] = f32_to_bf16(q * inv);

    float avail = (qt[n] <= t) ? 1.0f : 0.0f;
    #pragma unroll
    for (int m = 0; m < Mm - 1; ++m)
        memb[((size_t)n * Mm + m) * Dd + d] = memb[((size_t)n * Mm + m + 1) * Dd + d];
    memb[((size_t)n * Mm + Mm - 1) * Dd + d] = q * avail;
    __syncthreads();
    if (d < Mm - 1) validf[n * Mm + d] = vals[d + 1];
    if (d == Mm - 1) validf[n * Mm + d] = avail;
}

// ------------------------------------- feature row l2-normalize, f32 -> bf16
__global__ void normfeat_kernel(const float* __restrict__ f,
                                unsigned short* __restrict__ outv) {
    int p = blockIdx.x;
    int d = threadIdx.x;
    __shared__ float red[Dd];
    float v = f[(size_t)p * Dd + d];
    red[d] = v * v;
    __syncthreads();
    for (int s = Dd / 2; s > 0; s >>= 1) {
        if (d < s) red[d] += red[d + s];
        __syncthreads();
    }
    float inv = 1.0f / fmaxf(sqrtf(red[0]), 1e-12f);
    outv[(size_t)p * Dd + d] = f32_to_bf16(v * inv);
}

// ------------------------------------------------ bf16 WMMA correlation GEMM
// corr[m][p] = sum_k qn[m][k] * feat[p][k]
// Block = 256 threads = 8 waves; block tile = 16 queries x 256 pixels.
// Each wave owns a 16x32 output (two 16x16 accumulators sharing one A
// fragment) -> 16 v_wmma_f32_16x16x32_bf16 per wave over K=256, unpredicated
// (every launch uses Prows % 32 == 0, so a surviving wave always owns 32 cols).
// A tile (16x256 bf16) is staged to LDS by the Tensor Data Mover with
// descriptor padding (512B rows + 32B pad -> 272-half LDS stride).
__global__ void __launch_bounds__(256)
corr_wmma_kernel(const unsigned short* __restrict__ qn,
                 const unsigned short* __restrict__ feat,
                 float* __restrict__ corr,
                 int Prows) {
    __shared__ unsigned short As[16 * 272];   // padded stride: 272 halves (544 B)
    const int tid  = threadIdx.x;
    const int wave = tid >> 5;
    const int lane = tid & 31;
    const int m0   = blockIdx.x * 16;

#ifdef USE_TDM
    if (wave == 0) {
        unsigned long long ga = (unsigned long long)(size_t)(qn + (size_t)m0 * Dd);
        unsigned int       la = (unsigned int)(size_t)(&As[0]);
        // D# group 0: count=1 | lds_addr | global_addr[56:0] | type=2
        v4ui g0 = { 1u,
                    la,
                    (unsigned int)(ga & 0xFFFFFFFFull),
                    (unsigned int)((ga >> 32) & 0x01FFFFFFull) | (2u << 30) };
        // D# group 1:
        //  w0: data_size=1(2B)<<16 | pad_enable<<20 | pad_interval=6(128 dw)<<22
        //      | pad_amount=7(8 dw)<<25
        //  w1: tensor_dim0(=256)[15:0]<<16          (abar addr = 0)
        //  w2: tensor_dim1(=16)[15:0]<<16 | tensor_dim0[31:16](=0)
        //  w3: tile_dim0(=256)[15:0]<<16 | tensor_dim1[31:16](=0)
        //  w4: tile_dim1=16 | tile_dim2=0<<16
        //  w5: tensor_dim0_stride = 256
        //  w6,w7: 0
        v8ii g1 = { (int)0x0F910000, (int)0x01000000, (int)0x00100000,
                    (int)0x01000000, 16, 256, 0, 0 };
        v4ii gz = { 0, 0, 0, 0 };
#if __clang_major__ >= 23
        v8ii gz8 = { 0, 0, 0, 0, 0, 0, 0, 0 };
        __builtin_amdgcn_tensor_load_to_lds(g0, g1, gz, gz, gz8, 0);
#else
        __builtin_amdgcn_tensor_load_to_lds(g0, g1, gz, gz, 0);
#endif
        __builtin_amdgcn_s_wait_tensorcnt(0);
    }
    __syncthreads();
#else
    #pragma unroll
    for (int j = tid; j < 512; j += 256) {          // 512 chunks of 8 halves
        int row = j >> 5;
        int col = (j & 31) * 8;
        v8u val = *(const v8u*)(qn + (size_t)(m0 + row) * Dd + col);
        *(v8u*)(&As[row * 272 + col]) = val;
    }
    __syncthreads();
#endif

    const int p0 = blockIdx.y * 256 + wave * 32;
    if (p0 >= Prows) return;

    const int rsel = lane & 15;          // fragment row (A: M, B: N)
    const int kb   = (lane >> 4) << 3;   // K sub-offset per lane half

    const unsigned short* arow  = &As[rsel * 272];
    const unsigned short* brow0 = feat + (size_t)(p0 + rsel) * Dd;
    const unsigned short* brow1 = feat + (size_t)(p0 + 16 + rsel) * Dd;

    // speculative prefetch of the next pixel strip (gfx1250 global_prefetch_b8)
    if (p0 + 256 < Prows)
        __builtin_prefetch((const void*)(feat + (size_t)(p0 + 256 + rsel) * Dd), 0, 1);

    v8f acc0 = {}, acc1 = {};
    #pragma unroll
    for (int kk = 0; kk < Dd; kk += 32) {
        v8u alo = *(const v8u*)(arow + kk + kb);
        v8u ahi = *(const v8u*)(arow + kk + 16 + kb);
        v16u a16;
        #pragma unroll
        for (int i = 0; i < 8; ++i) { a16[i] = alo[i]; a16[i + 8] = ahi[i]; }
        v16bf a = __builtin_bit_cast(v16bf, a16);

        v8u blo0 = *(const v8u*)(brow0 + kk + kb);
        v8u bhi0 = *(const v8u*)(brow0 + kk + 16 + kb);
        v16u b16;
        #pragma unroll
        for (int i = 0; i < 8; ++i) { b16[i] = blo0[i]; b16[i + 8] = bhi0[i]; }
        acc0 = __builtin_amdgcn_wmma_f32_16x16x32_bf16(
            false, a, false, __builtin_bit_cast(v16bf, b16),
            (short)0, acc0, false, false);

        v8u blo1 = *(const v8u*)(brow1 + kk + kb);
        v8u bhi1 = *(const v8u*)(brow1 + kk + 16 + kb);
        v16u c16;
        #pragma unroll
        for (int i = 0; i < 8; ++i) { c16[i] = blo1[i]; c16[i + 8] = bhi1[i]; }
        acc1 = __builtin_amdgcn_wmma_f32_16x16x32_bf16(
            false, a, false, __builtin_bit_cast(v16bf, c16),
            (short)0, acc1, false, false);
    }

    // C layout: VGPR r -> M = r + 8*lane[4], N = lane[3:0]
    const int ncol = lane & 15;
    const int mhi  = (lane >> 4) * 8;
    #pragma unroll
    for (int r = 0; r < 8; ++r)
        corr[(size_t)(m0 + mhi + r) * Prows + p0 + ncol] = acc0[r];
    #pragma unroll
    for (int r = 0; r < 8; ++r)
        corr[(size_t)(m0 + mhi + r) * Prows + p0 + 16 + ncol] = acc1[r];
}

// --------------------------- bilinear upsample (half-pixel centers) of a map
__device__ __forceinline__ float upsample_map(const float* __restrict__ c,
                                              int n, int S, int h, int w) {
    float sy = (h + 0.5f) * ((float)S / 64.0f) - 0.5f;
    float sx = (w + 0.5f) * ((float)S / 64.0f) - 0.5f;
    float y0f = floorf(sy), x0f = floorf(sx);
    float wy = sy - y0f, wx = sx - x0f;
    int y0 = clampi((int)y0f,     0, S - 1);
    int y1 = clampi((int)y0f + 1, 0, S - 1);
    int x0 = clampi((int)x0f,     0, S - 1);
    int x1 = clampi((int)x0f + 1, 0, S - 1);
    const float* base = c + (size_t)n * S * S;
    return base[y0 * S + x0] * (1 - wx) * (1 - wy) + base[y0 * S + x1] * wx * (1 - wy)
         + base[y1 * S + x0] * (1 - wx) * wy       + base[y1 * S + x1] * wx * wy;
}

// ------------------------------------------- fuse 4 scales + bias, mask avail
__global__ void fuse_kernel(const float* __restrict__ c4,
                            const float* __restrict__ c8,
                            const float* __restrict__ c16,
                            const float* __restrict__ c32,
                            const float* __restrict__ wproj,
                            const float* __restrict__ bproj,
                            const int*   __restrict__ qt,
                            int t,
                            float* __restrict__ outp) {
    int idx = blockIdx.x * blockDim.x + threadIdx.x;   // over N*P
    int n = idx >> 12;            // / 4096
    int p = idx & (Pp - 1);
    int h = p / WF, w = p % WF;
    float avail = (qt[n] <= t) ? 1.0f : 0.0f;
    float r = wproj[0] * c4[(size_t)n * Pp + p];
    r += wproj[1] * upsample_map(c8,  n, 32, h, w);
    r += wproj[2] * upsample_map(c16, n, 16, h, w);
    r += wproj[3] * upsample_map(c32, n,  8, h, w);
    r += bproj[0];
    outp[((size_t)t * Nq + n) * Pp + p] = r * avail;
}

// ---------------------------------------------------------------------- host
extern "C" void kernel_launch(void* const* d_in, const int* in_sizes, int n_in,
                              void* d_out, int out_size, void* d_ws, size_t ws_size,
                              hipStream_t stream) {
    const float* f4   = (const float*)d_in[0];
    const float* f8   = (const float*)d_in[1];
    const float* f16  = (const float*)d_in[2];
    const float* f32  = (const float*)d_in[3];
    const float* ffu  = (const float*)d_in[4];
    const float* qxy  = (const float*)d_in[5];
    const float* wprj = (const float*)d_in[6];
    const float* bprj = (const float*)d_in[7];
    const int*   qt   = (const int*)d_in[8];
    float* outp = (float*)d_out;

    char* ws = (char*)d_ws;
    size_t off = 0;
    auto nextp = [&](size_t bytes) -> void* {
        off = (off + 255) & ~(size_t)255;
        void* p = ws + off;
        off += bytes;
        return p;
    };
    float* q_init = (float*)nextp((size_t)Nq * Dd * 4);
    float* memb   = (float*)nextp((size_t)Nq * Mm * Dd * 4);
    float* validf = (float*)nextp((size_t)Nq * Mm * 4);
    unsigned short* qn   = (unsigned short*)nextp((size_t)Nq * Dd * 2);
    unsigned short* fn4  = (unsigned short*)nextp((size_t)Pp * Dd * 2);
    unsigned short* fn8  = (unsigned short*)nextp((size_t)(Pp / 4) * Dd * 2);
    unsigned short* fn16 = (unsigned short*)nextp((size_t)(Pp / 16) * Dd * 2);
    unsigned short* fn32 = (unsigned short*)nextp((size_t)(Pp / 64) * Dd * 2);
    float* c4  = (float*)nextp((size_t)Nq * Pp * 4);
    float* c8  = (float*)nextp((size_t)Nq * (Pp / 4) * 4);
    float* c16 = (float*)nextp((size_t)Nq * (Pp / 16) * 4);
    float* c32 = (float*)nextp((size_t)Nq * (Pp / 64) * 4);

    zero_state_kernel<<<(Nq * Mm * Dd + 255) / 256, 256, 0, stream>>>(memb, validf);
    grid_sample_kernel<<<Nq, Dd, 0, stream>>>(ffu, qxy, qt, q_init);

    for (int t = 0; t < Tx; ++t) {
        qupdate_kernel<<<Nq, Dd, 0, stream>>>(q_init, memb, validf, qt, t, qn);

        normfeat_kernel<<<Pp,      Dd, 0, stream>>>(f4  + (size_t)t * Pp * Dd,        fn4);
        normfeat_kernel<<<Pp / 4,  Dd, 0, stream>>>(f8  + (size_t)t * (Pp / 4) * Dd,  fn8);
        normfeat_kernel<<<Pp / 16, Dd, 0, stream>>>(f16 + (size_t)t * (Pp / 16) * Dd, fn16);
        normfeat_kernel<<<Pp / 64, Dd, 0, stream>>>(f32 + (size_t)t * (Pp / 64) * Dd, fn32);

        corr_wmma_kernel<<<dim3(Nq / 16, (Pp      + 255) / 256), 256, 0, stream>>>(qn, fn4,  c4,  Pp);
        corr_wmma_kernel<<<dim3(Nq / 16, (Pp / 4  + 255) / 256), 256, 0, stream>>>(qn, fn8,  c8,  Pp / 4);
        corr_wmma_kernel<<<dim3(Nq / 16, (Pp / 16 + 255) / 256), 256, 0, stream>>>(qn, fn16, c16, Pp / 16);
        corr_wmma_kernel<<<dim3(Nq / 16, (Pp / 64 + 255) / 256), 256, 0, stream>>>(qn, fn32, c32, Pp / 64);

        fuse_kernel<<<(Nq * Pp) / 256, 256, 0, stream>>>(c4, c8, c16, c32,
                                                         wprj, bprj, qt, t, outp);
    }
}
